// LlamaAttention_36266703848200
// MI455X (gfx1250) — matrix operand order repaired and checked by reading, not compiled, not run on hardware
//
#include <hip/hip_runtime.h>

typedef _Float16 h16;
typedef __attribute__((ext_vector_type(8)))  _Float16 v8h;
typedef __attribute__((ext_vector_type(16))) _Float16 v16h;
typedef __attribute__((ext_vector_type(8)))  float    v8f;

// ---------------------------------------------------------------------------
// WMMA helper: D = A(16x32 f16) x B(32x16 f16) + C(16x16 f32), wave32
// ---------------------------------------------------------------------------
static __device__ __forceinline__ v8f wmma16(v16h a, v16h b, v8f c) {
  return __builtin_amdgcn_wmma_f32_16x16x32_f16(
      /*neg_a=*/false, a, /*neg_b=*/false, b,
      /*c_mod=*/(short)0, c, /*reuse_a=*/false, /*reuse_b=*/false);
}

static __device__ __forceinline__ v16h cat16(v8h a, v8h b) {
  return __builtin_shufflevector(a, b, 0,1,2,3,4,5,6,7,8,9,10,11,12,13,14,15);
}

// 16 f16 from two contiguous 8-half chunks (works for global or LDS pointers)
static __device__ __forceinline__ v16h load16h(const h16* p0, const h16* p1) {
  return cat16(*(const v8h*)p0, *(const v8h*)p1);
}

// 16 f16 converted from two contiguous 8-float chunks
static __device__ __forceinline__ v16h load16f(const float* p0, const float* p1) {
  float4 a0 = *(const float4*)p0;
  float4 a1 = *(const float4*)(p0 + 4);
  float4 b0 = *(const float4*)p1;
  float4 b1 = *(const float4*)(p1 + 4);
  v16h r;
  r[0]=(h16)a0.x;  r[1]=(h16)a0.y;  r[2]=(h16)a0.z;  r[3]=(h16)a0.w;
  r[4]=(h16)a1.x;  r[5]=(h16)a1.y;  r[6]=(h16)a1.z;  r[7]=(h16)a1.w;
  r[8]=(h16)b0.x;  r[9]=(h16)b0.y;  r[10]=(h16)b0.z; r[11]=(h16)b0.w;
  r[12]=(h16)b1.x; r[13]=(h16)b1.y; r[14]=(h16)b1.z; r[15]=(h16)b1.w;
  return r;
}

// A-fragment (16x32, 16-bit) per CDNA5 ISA layout:
//   lane L: m = L&15, a[0..7]  = A[m][k0 + (L>=16?8:0) + 0..7]
//           a[8..15] = A[m][k0 + 16 + (L>=16?8:0) + 0..7]
static __device__ __forceinline__ v16h load_afrag(const float* row, int hh) {
  return load16f(row + hh * 8, row + 16 + hh * 8);
}
static __device__ __forceinline__ v16h load_afrag(const h16* row, int hh) {
  return load16h(row + hh * 8, row + 16 + hh * 8);
}

// ---------------------------------------------------------------------------
// CDNA5 async global->LDS copy (ASYNCcnt-tracked, bypasses VGPRs).
// lds_off = LDS byte address (low 32 bits of the generic pointer, per the
// aperture mapping).
// ---------------------------------------------------------------------------
static __device__ __forceinline__ void async_copy_b128(void* lds_dst,
                                                       const void* gsrc) {
  unsigned lds_off = (unsigned)(unsigned long long)lds_dst;
  asm volatile("global_load_async_to_lds_b128 %0, %1, off"
               :: "v"(lds_off), "v"(gsrc)
               : "memory");
}
static __device__ __forceinline__ void async_wait0() {
  asm volatile("s_wait_asynccnt 0x0" ::: "memory");
}

// ---------------------------------------------------------------------------
// GEMM: out[M][N] = X[M][K] * W[N][K]^T     (row-major everywhere)
// Block: 128 threads = 4 waves. Block tile 64(M) x 64(N), K-step 32.
// W tile staged into LDS as f16; A fragments loaded straight from global.
// ---------------------------------------------------------------------------
template <typename XT, typename OT>
__global__ __launch_bounds__(128)
void gemm_xwt(const XT* __restrict__ X, const float* __restrict__ W,
              OT* __restrict__ out, int M, int N, int K) {
  __shared__ h16 Wbuf[64][48];  // 64 N-rows x 32 K-cols (+16 pad halves)

  const int tid  = threadIdx.x;
  const int lane = tid & 31;
  const int w    = tid >> 5;
  const int hh   = (lane >> 4) & 1;  // lane half (wave32)
  const int ln   = lane & 15;
  const int bm   = blockIdx.x * 64;
  const int bn   = blockIdx.y * 64;

  v8f acc[4] = {};

  const XT* Xrow = X + (size_t)(bm + w * 16 + ln) * K;

  // cooperative W staging: thread -> (row wr, 16 cols at wc)
  const int wr = tid >> 1;
  const int wc = (tid & 1) * 16;
  const float* Wst = W + (size_t)(bn + wr) * K + wc;

  for (int k0 = 0; k0 < K; k0 += 32) {
    __syncthreads();
    {
      const float* src = Wst + k0;
      h16* dst = &Wbuf[wr][wc];
      #pragma unroll
      for (int i = 0; i < 16; i += 4) {
        float4 t = *(const float4*)(src + i);
        dst[i + 0] = (h16)t.x; dst[i + 1] = (h16)t.y;
        dst[i + 2] = (h16)t.z; dst[i + 3] = (h16)t.w;
      }
    }
    __syncthreads();

    if (k0 + 32 < K) __builtin_prefetch(Xrow + k0 + 32, 0, 1);

    v16h af = load_afrag(Xrow + k0, hh);
    #pragma unroll
    for (int ct = 0; ct < 4; ct++) {
      // B[r][n] = W[bn+ct*16+n][k0+r]; lane: n=ln, b[0..7]=Wbuf[ct*16+ln][hh*8+t], b[8..15]=Wbuf[ct*16+ln][16+hh*8+t]
      v16h bf = load16h(&Wbuf[ct * 16 + ln][hh * 8],
                        &Wbuf[ct * 16 + ln][16 + hh * 8]);
      acc[ct] = wmma16(af, bf, acc[ct]);
    }
  }

  // C layout: row m = p + 8*hh (VGPR p), col n = ln
  #pragma unroll
  for (int ct = 0; ct < 4; ct++) {
    #pragma unroll
    for (int p = 0; p < 8; p++) {
      const int r = bm + w * 16 + p + hh * 8;
      const int c = bn + ct * 16 + ln;
      out[(size_t)r * N + c] = (OT)acc[ct][p];
    }
  }
}

// ---------------------------------------------------------------------------
// RoPE (in-place on f16 tensor laid out [2048][nheads*128])
// ---------------------------------------------------------------------------
__global__ void rope_kernel(h16* __restrict__ T, const int* __restrict__ pos_ids,
                            int nheads) {
  const int idx   = blockIdx.x * blockDim.x + threadIdx.x;
  const int total = 2048 * nheads * 64;
  if (idx >= total) return;
  const int d = idx & 63;
  const int t = idx >> 6;
  const int h = t % nheads;
  const int s = t / nheads;

  const float pos = (float)pos_ids[s];
  // inv_freq[d] = theta^(-d/64), theta = 500000 -> exp(-d * ln(theta)/64)
  const float invf = __expf(-(float)d * 0.20503692777194267f);
  const float f = pos * invf;
  float sn, cs;
  sincosf(f, &sn, &cs);

  h16* base = T + (size_t)s * (nheads * 128) + h * 128 + d;
  const float a = (float)base[0];
  const float b = (float)base[64];
  base[0]  = (h16)(a * cs - b * sn);
  base[64] = (h16)(b * cs + a * sn);
}

// ---------------------------------------------------------------------------
// Flash attention, causal. Block = 128 threads (4 waves) on one head,
// 64 q-rows (wave w -> 16-row q tile). 32 keys per step, online softmax.
// K tile staged via CDNA5 async global->LDS; V staged transposed via VGPRs.
// ---------------------------------------------------------------------------
__global__ __launch_bounds__(128)
void flash_attn(const h16* __restrict__ Q, const h16* __restrict__ Km,
                const h16* __restrict__ Vm, h16* __restrict__ O) {
  __shared__ h16 Kbuf[32][136];     // [key][dim] + pad
  __shared__ h16 VT[128][40];       // [dim][key] (transposed) + pad
  __shared__ h16 Pbuf[4][16][32];   // per-wave P relayout buffer

  const int tid  = threadIdx.x;
  const int lane = tid & 31;
  const int w    = tid >> 5;
  const int hh   = (lane >> 4) & 1;
  const int ln   = lane & 15;
  const int qb   = blockIdx.x;
  const int head = blockIdx.y;
  const int kvh  = head >> 2;             // 32 heads -> 8 KV heads
  const int q0   = qb * 64 + w * 16;      // wave's q-tile start

  // Q fragments: 4 k-steps over head_dim=128, straight from global f16
  const h16* Qrow = Q + (size_t)(q0 + ln) * 4096 + head * 128;
  v16h qf[4];
  #pragma unroll
  for (int kk = 0; kk < 4; kk++)
    qf[kk] = load16h(Qrow + kk * 32 + hh * 8, Qrow + kk * 32 + 16 + hh * 8);

  v8f acc[8] = {};
  float mrow[8], lrow[8];
  #pragma unroll
  for (int p = 0; p < 8; p++) { mrow[p] = -3.0e38f; lrow[p] = 0.f; }

  const int nsteps = qb * 2 + 2;  // keys up to q0_block+63, uniform per block
  const float scl = 0.08838834764831845f;  // 1/sqrt(128)

  const int sr = tid >> 2;         // staging: key row 0..31
  const int sc = (tid & 3) * 32;   // staging: 32 dims starting here

  for (int ks = 0; ks < nsteps; ks++) {
    const int kbase = ks * 32;
    __syncthreads();
    {
      const h16* ksrc = Km + (size_t)(kbase + sr) * 1024 + kvh * 128 + sc;
      const h16* vsrc = Vm + (size_t)(kbase + sr) * 1024 + kvh * 128 + sc;

      // K tile: pure byte copy -> async DMA into LDS (ASYNCcnt)
      #pragma unroll
      for (int i = 0; i < 32; i += 8)
        async_copy_b128(&Kbuf[sr][sc + i], ksrc + i);

      // V tile: b128 global loads, transposed scalar stores into LDS
      #pragma unroll
      for (int i = 0; i < 32; i += 8) {
        v8h vv = *(const v8h*)(vsrc + i);
        #pragma unroll
        for (int j = 0; j < 8; j++) VT[sc + i + j][sr] = vv[j];
      }

      async_wait0();   // our async K copies are done (cross-wave via barrier)
    }
    __syncthreads();

    // scores: S[m][n] = sum_d Q[m][d] * K[key n][d]; two 16-key tiles
    v8f s0 = {}, s1 = {};
    #pragma unroll
    for (int kk = 0; kk < 4; kk++) {
      v16h b0 = load16h(&Kbuf[ln][kk * 32 + hh * 8],
                        &Kbuf[ln][kk * 32 + 16 + hh * 8]);
      v16h b1 = load16h(&Kbuf[16 + ln][kk * 32 + hh * 8],
                        &Kbuf[16 + ln][kk * 32 + 16 + hh * 8]);
      s0 = wmma16(qf[kk], b0, s0);
      s1 = wmma16(qf[kk], b1, s1);
    }

    // online softmax: rows live per (VGPR p, lane half); reduce across 16 lanes
    float e0[8], e1[8], scrow[8];
    #pragma unroll
    for (int p = 0; p < 8; p++) {
      const int qg = q0 + p + hh * 8;
      float a0 = (kbase + ln <= qg)      ? s0[p] * scl : -1.0e30f;
      float a1 = (kbase + 16 + ln <= qg) ? s1[p] * scl : -1.0e30f;
      float t = fmaxf(a0, a1);
      #pragma unroll
      for (int off = 1; off < 16; off <<= 1)
        t = fmaxf(t, __shfl_xor(t, off, 32));
      const float mn = fmaxf(mrow[p], t);
      const float rs = __expf(mrow[p] - mn);
      mrow[p] = mn;
      e0[p] = __expf(a0 - mn);
      e1[p] = __expf(a1 - mn);
      float su = e0[p] + e1[p];
      #pragma unroll
      for (int off = 1; off < 16; off <<= 1)
        su += __shfl_xor(su, off, 32);
      lrow[p] = lrow[p] * rs + su;
      scrow[p] = rs;
    }

    #pragma unroll
    for (int dt = 0; dt < 8; dt++)
      #pragma unroll
      for (int p = 0; p < 8; p++)
        acc[dt][p] *= scrow[p];

    // P: C-layout -> A-layout via per-wave LDS buffer
    #pragma unroll
    for (int p = 0; p < 8; p++) {
      Pbuf[w][p + hh * 8][ln]      = (h16)e0[p];
      Pbuf[w][p + hh * 8][16 + ln] = (h16)e1[p];
    }
    __syncthreads();

    v16h pf = load16h(&Pbuf[w][ln][hh * 8], &Pbuf[w][ln][16 + hh * 8]);
    #pragma unroll
    for (int dt = 0; dt < 8; dt++) {
      // B[k][c] = V[key k][dim 16*dt+c] = VT[dim][k]; table order over k
      v16h bf = load16h(&VT[dt * 16 + ln][hh * 8],
                        &VT[dt * 16 + ln][16 + hh * 8]);
      acc[dt] = wmma16(pf, bf, acc[dt]);
    }
  }

  #pragma unroll
  for (int p = 0; p < 8; p++) lrow[p] = 1.0f / lrow[p];
  #pragma unroll
  for (int dt = 0; dt < 8; dt++)
    #pragma unroll
    for (int p = 0; p < 8; p++)
      O[(size_t)(q0 + p + hh * 8) * 4096 + head * 128 + dt * 16 + ln] =
          (h16)(acc[dt][p] * lrow[p]);
}

// ---------------------------------------------------------------------------
// Launch: QKV proj -> RoPE -> flash attention -> O proj
// ---------------------------------------------------------------------------
extern "C" void kernel_launch(void* const* d_in, const int* in_sizes, int n_in,
                              void* d_out, int out_size, void* d_ws, size_t ws_size,
                              hipStream_t stream) {
  const float* X  = (const float*)d_in[0];   // [2048][4096]
  const float* Wq = (const float*)d_in[1];   // [4096][4096]
  const float* Wk = (const float*)d_in[2];   // [1024][4096]
  const float* Wv = (const float*)d_in[3];   // [1024][4096]
  const float* Wo = (const float*)d_in[4];   // [4096][4096]
  const int*  pos = (const int*)d_in[5];     // [2048]
  float* out = (float*)d_out;

  // workspace (f16): Q 16MB, K 4MB, V 4MB, attn-out 16MB
  h16* Qh = (h16*)d_ws;
  h16* Kh = Qh + (size_t)2048 * 4096;
  h16* Vh = Kh + (size_t)2048 * 1024;
  h16* AO = Vh + (size_t)2048 * 1024;

  const dim3 blk(128);

  gemm_xwt<float, h16><<<dim3(32, 64), blk, 0, stream>>>(X, Wq, Qh, 2048, 4096, 4096);
  gemm_xwt<float, h16><<<dim3(32, 16), blk, 0, stream>>>(X, Wk, Kh, 2048, 1024, 4096);
  gemm_xwt<float, h16><<<dim3(32, 16), blk, 0, stream>>>(X, Wv, Vh, 2048, 1024, 4096);

  {
    const int totalQ = 2048 * 32 * 64;
    rope_kernel<<<(totalQ + 255) / 256, 256, 0, stream>>>(Qh, pos, 32);
    const int totalK = 2048 * 8 * 64;
    rope_kernel<<<(totalK + 255) / 256, 256, 0, stream>>>(Kh, pos, 8);
  }

  flash_attn<<<dim3(32, 32), blk, 0, stream>>>(Qh, Kh, Vh, AO);

  gemm_xwt<h16, float><<<dim3(32, 64), blk, 0, stream>>>(AO, Wo, out, 2048, 4096, 4096);
}
